// SelfAttention_13872744366524
// MI455X (gfx1250) — matrix operand order repaired and checked by reading, not compiled, not run on hardware
//
#include <hip/hip_runtime.h>
#include <math.h>

typedef __attribute__((ext_vector_type(16))) __bf16 v16bf;
typedef __attribute__((ext_vector_type(8)))  float  v8f;

#define BQ 2
#define SQ 2048
#define HIDQ 2048
#define HQ 16
#define KVQ 4
#define DQ 128
#define GQ (HQ / KVQ)

__device__ __forceinline__ v8f v8f_zero() {
  v8f v;
#pragma unroll
  for (int i = 0; i < 8; ++i) v[i] = 0.f;
  return v;
}

// D = A(16x32 bf16) * B(32x16 bf16) + C(16x16 f32)   [wave32]
__device__ __forceinline__ v8f wmma_bf16(v16bf a, v16bf b, v8f c) {
  return __builtin_amdgcn_wmma_f32_16x16x32_bf16(
      /*neg_a=*/false, a, /*neg_b=*/false, b,
      /*c_mod=*/(short)0, c, /*reuse_a=*/false, /*reuse_b=*/false);
}

// ---------------------------------------------------------------------------
// Generic C[M,N] = A[M,K] @ W[K,N], f32 in / f32 out, bf16 WMMA inner product.
// Optional elementwise gate on A:  A_eff = A * sigmoid(Gate).
// Block: 256 threads (8 waves). Tile: 128x128, K-step 32.
// Wave grid 4(m) x 2(n); each wave: 2x4 tiles of 16x16 -> 8 accumulators.
// ---------------------------------------------------------------------------
template <bool GATED>
__global__ void __launch_bounds__(256)
gemm_bf16(const float* __restrict__ A, int lda,
          const float* __restrict__ Gate, int ldg,
          const float* __restrict__ Bw,
          float* __restrict__ C,
          int M, int N, int K) {
  constexpr int BM = 128, BN = 128, BK = 32, BKP = 40; // padded LDS row
  __shared__ __bf16 sA[BM * BKP];
  __shared__ __bf16 sB[BN * BKP]; // transposed: [n][k]

  const int tid  = threadIdx.x;
  const int lane = tid & 31;
  const int wave = tid >> 5;
  const int wm = wave & 3;   // 0..3  -> 32-row slab
  const int wn = wave >> 2;  // 0..1  -> 64-col slab
  const int m0 = blockIdx.y * BM;
  const int n0 = blockIdx.x * BN;
  const int hl  = lane >> 4; // half-wave id
  const int l16 = lane & 15;

  v8f acc[2][4];
#pragma unroll
  for (int i = 0; i < 2; ++i)
#pragma unroll
    for (int j = 0; j < 4; ++j) acc[i][j] = v8f_zero();

  for (int k0 = 0; k0 < K; k0 += BK) {
    // prefetch next K-slab (speculative, no counter cost)
    if (k0 + BK < K) {
      __builtin_prefetch(&A[(size_t)(m0 + (tid >> 1)) * lda + k0 + BK], 0, 1);
      __builtin_prefetch(&Bw[(size_t)(k0 + BK + (tid >> 3)) * N + n0], 0, 1);
    }
    // stage A tile (f32 -> bf16), optionally gated
    for (int idx = tid; idx < BM * BK; idx += 256) {
      int r = idx >> 5, c = idx & 31;
      float v = A[(size_t)(m0 + r) * lda + (k0 + c)];
      if (GATED) {
        float g = Gate[(size_t)(m0 + r) * ldg + (k0 + c)];
        v *= 1.f / (1.f + __expf(-g));
      }
      sA[r * BKP + c] = (__bf16)v;
    }
    // stage B tile transposed: sB[n][k]
    for (int idx = tid; idx < BK * BN; idx += 256) {
      int r = idx >> 7, c = idx & 127; // r = k, c = n
      sB[c * BKP + r] = (__bf16)Bw[(size_t)(k0 + r) * N + (n0 + c)];
    }
    __syncthreads();

    v16bf afr[2], bfr[4];
#pragma unroll
    for (int mm = 0; mm < 2; ++mm) { // A frag: row = l16, K = 16*hl + i
      const __bf16* p = &sA[(wm * 32 + mm * 16 + l16) * BKP + hl * 16];
#pragma unroll
      for (int i = 0; i < 8; ++i) afr[mm][i] = p[i];
#pragma unroll
      for (int i = 0; i < 8; ++i) afr[mm][8 + i] = p[8 + i];
    }
#pragma unroll
    for (int nn = 0; nn < 4; ++nn) { // B frag: N = l16, K = 16*hl + i
      const __bf16* p = &sB[(wn * 64 + nn * 16 + l16) * BKP + hl * 16];
#pragma unroll
      for (int i = 0; i < 16; ++i) bfr[nn][i] = p[i];
    }
#pragma unroll
    for (int mm = 0; mm < 2; ++mm)
#pragma unroll
      for (int nn = 0; nn < 4; ++nn)
        acc[mm][nn] = wmma_bf16(afr[mm], bfr[nn], acc[mm][nn]);
    __syncthreads();
  }

#pragma unroll
  for (int mm = 0; mm < 2; ++mm)
#pragma unroll
    for (int nn = 0; nn < 4; ++nn)
#pragma unroll
      for (int r = 0; r < 8; ++r) {
        int row = m0 + wm * 32 + mm * 16 + r + 8 * hl;
        int col = n0 + wn * 64 + nn * 16 + l16;
        C[(size_t)row * N + col] = acc[mm][nn][r];
      }
}

// ---------------------------------------------------------------------------
// RMSNorm (+ (1+w) scale) then RoPE, per (token, head). Block = 128 = D.
// src row layout: [token][src_stride] with head slice at head*D.
// dst: [token][nheads][D].
// ---------------------------------------------------------------------------
__global__ void __launch_bounds__(128)
rmsnorm_rope(const float* __restrict__ src, int src_stride,
             float* __restrict__ dst, int nheads,
             const float* __restrict__ w,
             const float* __restrict__ cs,
             const float* __restrict__ sn) {
  const int m = blockIdx.x;  // token in [0, B*S)
  const int h = blockIdx.y;
  const int d = threadIdx.x; // 0..127
  __shared__ float red[4];
  __shared__ float buf[DQ];

  float x = src[(size_t)m * src_stride + h * DQ + d];
  float ss = x * x;
#pragma unroll
  for (int o = 1; o < 32; o <<= 1) ss += __shfl_xor(ss, o);
  if ((threadIdx.x & 31) == 0) red[threadIdx.x >> 5] = ss;
  __syncthreads();
  float tot = red[0] + red[1] + red[2] + red[3];
  float xn = x * rsqrtf(tot * (1.f / DQ) + 1e-6f) * (1.f + w[d]);
  buf[d] = xn;
  __syncthreads();
  float other = buf[d ^ 64];
  float rot = (d < 64) ? -other : other; // rotate_half
  float c = cs[(size_t)m * DQ + d];
  float s = sn[(size_t)m * DQ + d];
  dst[((size_t)m * nheads + h) * DQ + d] = xn * c + rot * s;
}

// V: [B,S,KV,D] -> Vt: [B,KV,D,S]  (so P@V B-fragments read contiguous keys)
__global__ void __launch_bounds__(256)
vtrans(const float* __restrict__ V, float* __restrict__ Vt) {
  int idx = blockIdx.x * blockDim.x + threadIdx.x; // B*S*KV*D = 2^21 exact
  int d  = idx & (DQ - 1);
  int kv = (idx >> 7) & (KVQ - 1);
  int s  = (idx >> 9) & (SQ - 1);
  int b  = idx >> 20;
  Vt[(((size_t)b * KVQ + kv) * DQ + d) * SQ + s] = V[idx];
}

// ---------------------------------------------------------------------------
// Causal flash attention, GQA (G=4). Block = 8 waves; each wave owns 16 query
// rows; block covers 128 rows of one (b,h). Online softmax, bf16 WMMA for
// both QK^T and P@V. Scores never leave registers/LDS.
// ---------------------------------------------------------------------------
__global__ void __launch_bounds__(256)
flash_attn(const float* __restrict__ Q,   // [B*S, H, D]  (normed+roped)
           const float* __restrict__ Kk,  // [B*S, KV, D] (normed+roped)
           const float* __restrict__ Vt,  // [B, KV, D, S]
           float* __restrict__ O) {       // [B*S, H, D]
  const int bh = blockIdx.y;
  const int b = bh / HQ;
  const int h = bh % HQ;
  const int kv = h / GQ;
  const int lane = threadIdx.x & 31;
  const int wave = threadIdx.x >> 5;
  const int hl  = lane >> 4;
  const int l16 = lane & 15;
  const int qr0 = blockIdx.x * 128 + wave * 16;

  __shared__ __bf16 plds[8][16][32]; // per-wave private P tile

  const float scale = 0.08838834764831845f; // D^-0.5, folded into Q
  v16bf qf[4];
  {
    const float* qp = Q + ((size_t)(b * SQ + qr0 + l16) * HQ + h) * DQ;
#pragma unroll
    for (int dc = 0; dc < 4; ++dc) {
      int klo = dc * 32 + hl * 16;
#pragma unroll
      for (int i = 0; i < 8; ++i) qf[dc][i] = (__bf16)(qp[klo + i] * scale);
#pragma unroll
      for (int i = 0; i < 8; ++i) qf[dc][8 + i] = (__bf16)(qp[klo + 8 + i] * scale);
    }
  }

  v8f acc[8]; // 16 rows x 128 d, f32
#pragma unroll
  for (int t = 0; t < 8; ++t) acc[t] = v8f_zero();
  float mrow[8], lrow[8];
#pragma unroll
  for (int r = 0; r < 8; ++r) { mrow[r] = -3.0e38f; lrow[r] = 0.f; }

  for (int j0 = 0; j0 < qr0 + 16; j0 += 32) { // causal KV chunks of 32
    // --- scores: two 16-key tiles, K-dim = D via 4 chained WMMAs each ---
    v8f sc[2];
#pragma unroll
    for (int n = 0; n < 2; ++n) {
      v8f c = v8f_zero();
      const float* kp =
          Kk + ((size_t)(b * SQ + j0 + n * 16 + l16) * KVQ + kv) * DQ + hl * 16;
#pragma unroll
      for (int dc = 0; dc < 4; ++dc) {
        v16bf kf;
#pragma unroll
        for (int i = 0; i < 16; ++i) kf[i] = (__bf16)kp[dc * 32 + i];
        c = wmma_bf16(qf[dc], kf, c);
      }
      sc[n] = c;
    }
    // --- causal mask + online softmax (row stats across 16-lane halves) ---
#pragma unroll
    for (int r = 0; r < 8; ++r) {
      int qrow = qr0 + r + 8 * hl;
      float s0 = sc[0][r] + ((j0 + l16 > qrow) ? -1.0e9f : 0.f);
      float s1 = sc[1][r] + ((j0 + 16 + l16 > qrow) ? -1.0e9f : 0.f);
      float mx = fmaxf(s0, s1);
      mx = fmaxf(mx, __shfl_xor(mx, 1));
      mx = fmaxf(mx, __shfl_xor(mx, 2));
      mx = fmaxf(mx, __shfl_xor(mx, 4));
      mx = fmaxf(mx, __shfl_xor(mx, 8));
      float mnew = fmaxf(mrow[r], mx);
      float corr = __expf(mrow[r] - mnew);
      float p0 = __expf(s0 - mnew);
      float p1 = __expf(s1 - mnew);
      float rs = p0 + p1;
      rs += __shfl_xor(rs, 1);
      rs += __shfl_xor(rs, 2);
      rs += __shfl_xor(rs, 4);
      rs += __shfl_xor(rs, 8);
      lrow[r] = lrow[r] * corr + rs;
      mrow[r] = mnew;
#pragma unroll
      for (int t = 0; t < 8; ++t) acc[t][r] *= corr;
      // C-layout -> LDS tile (for transpose into A-layout)
      plds[wave][r + 8 * hl][l16]      = (__bf16)p0;
      plds[wave][r + 8 * hl][16 + l16] = (__bf16)p1;
    }
    // Intra-wave RAW on per-wave-private LDS; DS is in-order per wave.
    asm volatile("s_wait_dscnt 0x0" ::: "memory");
    v16bf pf; // P as 16x32 A fragment
    {
      const __bf16* pp = &plds[wave][l16][hl * 16];
#pragma unroll
      for (int i = 0; i < 8; ++i) pf[i] = pp[i];
#pragma unroll
      for (int i = 0; i < 8; ++i) pf[8 + i] = pp[8 + i];
    }
    // --- P @ V : 8 d-tiles of 16 ---
    const float* vp = Vt + (((size_t)b * KVQ + kv) * DQ) * SQ + j0 + hl * 16;
#pragma unroll
    for (int t = 0; t < 8; ++t) {
      v16bf vf;
      const float* vq = vp + (size_t)(t * 16 + l16) * SQ;
#pragma unroll
      for (int i = 0; i < 16; ++i) vf[i] = (__bf16)vq[i];
      acc[t] = wmma_bf16(pf, vf, acc[t]);
    }
    asm volatile("s_wait_dscnt 0x0" ::: "memory"); // WAR before next chunk
  }

#pragma unroll
  for (int t = 0; t < 8; ++t)
#pragma unroll
    for (int r = 0; r < 8; ++r) {
      int qrow = qr0 + r + 8 * hl;
      O[((size_t)(b * SQ + qrow) * HQ + h) * DQ + t * 16 + l16] =
          acc[t][r] / lrow[r];
    }
}

// ---------------------------------------------------------------------------
extern "C" void kernel_launch(void* const* d_in, const int* in_sizes, int n_in,
                              void* d_out, int out_size, void* d_ws, size_t ws_size,
                              hipStream_t stream) {
  (void)in_sizes; (void)n_in; (void)out_size; (void)ws_size;
  const float* X  = (const float*)d_in[0];
  const float* cs = (const float*)d_in[1];
  const float* sn = (const float*)d_in[2];
  // d_in[3] attention_mask: causal, applied analytically in flash_attn
  const float* Wq = (const float*)d_in[4];
  const float* Wk = (const float*)d_in[5];
  const float* Wv = (const float*)d_in[6];
  const float* Wo = (const float*)d_in[7];
  const float* qw = (const float*)d_in[8];
  const float* kw = (const float*)d_in[9];
  float* out = (float*)d_out;

  char* ws = (char*)d_ws;
  const int M = BQ * SQ; // 4096 tokens
  // workspace map (160 MB total)
  float* qg    = (float*)(ws);                        // [M, 2*H*D] = 64 MB
  float* kproj = (float*)(ws + ((size_t)64 << 20));   // [M, KV*D]  =  8 MB
  float* vproj = (float*)(ws + ((size_t)72 << 20));   // [M, KV*D]  =  8 MB
  float* Qr    = (float*)(ws + ((size_t)80 << 20));   // [M, H, D]  = 32 MB
  float* Kr    = (float*)(ws + ((size_t)112 << 20));  // [M, KV, D] =  8 MB
  float* Vt    = (float*)(ws + ((size_t)120 << 20));  // [B,KV,D,S] =  8 MB
  float* attn  = (float*)(ws + ((size_t)128 << 20));  // [M, H*D]   = 32 MB

  dim3 blk(256);
  // projections
  gemm_bf16<false><<<dim3((2 * HQ * DQ) / 128, M / 128), blk, 0, stream>>>(
      X, HIDQ, nullptr, 0, Wq, qg, M, 2 * HQ * DQ, HIDQ);
  gemm_bf16<false><<<dim3((KVQ * DQ) / 128, M / 128), blk, 0, stream>>>(
      X, HIDQ, nullptr, 0, Wk, kproj, M, KVQ * DQ, HIDQ);
  gemm_bf16<false><<<dim3((KVQ * DQ) / 128, M / 128), blk, 0, stream>>>(
      X, HIDQ, nullptr, 0, Wv, vproj, M, KVQ * DQ, HIDQ);
  // qk-norm + rope
  rmsnorm_rope<<<dim3(M, HQ), dim3(128), 0, stream>>>(qg, 2 * HQ * DQ, Qr, HQ, qw, cs, sn);
  rmsnorm_rope<<<dim3(M, KVQ), dim3(128), 0, stream>>>(kproj, KVQ * DQ, Kr, KVQ, kw, cs, sn);
  // V transpose for coalesced P@V fragments
  vtrans<<<dim3((BQ * SQ * KVQ * DQ) / 256), dim3(256), 0, stream>>>(vproj, Vt);
  // flash attention
  flash_attn<<<dim3(SQ / 128, BQ * HQ), blk, 0, stream>>>(Qr, Kr, Vt, attn);
  // gated output projection: out = (attn * sigmoid(gate)) @ Wo
  gemm_bf16<true><<<dim3(HIDQ / 128, M / 128), blk, 0, stream>>>(
      attn, HQ * DQ, qg + HQ * DQ, 2 * HQ * DQ, Wo, out, M, HIDQ, HIDQ);
}
